// Block_63187558858832
// MI455X (gfx1250) — matrix operand order repaired and checked by reading, not compiled, not run on hardware
//
#include <hip/hip_runtime.h>
#include <hip/hip_bf16.h>

typedef __bf16 v16bf __attribute__((ext_vector_type(16)));
typedef float  v8f   __attribute__((ext_vector_type(8)));

union FragU { float4 f[2]; v16bf v; };

// A-matrix 16x32 bf16 fragment (per ISA table):
// lanes 0-15: K = kh..kh+7 and 16+kh..23+kh with kh=0 ; lanes 16-31: kh=8
__device__ __forceinline__ v16bf load_fragA(const __bf16* rowPtr, int kh) {
  FragU u;
  u.f[0] = *(const float4*)(rowPtr + kh);
  u.f[1] = *(const float4*)(rowPtr + 16 + kh);
  return u.v;
}
// B-matrix 32x16 bf16 fragment: lanes 0-15 hold K=0..7,16..23, lanes 16-31 hold K=8..15,24..31
// colPtr points at K-contiguous storage for this column.
__device__ __forceinline__ v16bf load_fragB(const __bf16* colPtr, int hi) {
  FragU u;
  u.f[0] = *(const float4*)(colPtr + hi * 8);
  u.f[1] = *(const float4*)(colPtr + 16 + hi * 8);
  return u.v;
}

// ---------------------------------------------------------------------------
// Generic bf16 WMMA GEMM: C[M,N] = A[M,K] * B[K,N] (+bias)(+res)(relu)
// Block tile 128x64, 8 waves (4x2), each wave 32x32 (2x2 wmma tiles), BK=32.
// Register-prefetch pipeline: next K-tile global loads issued before compute.
// ---------------------------------------------------------------------------
template<bool RELU, bool BF16OUT, bool HAS_BIAS, bool HAS_RES>
__global__ __launch_bounds__(256)
void gemm_wmma(const __bf16* __restrict__ A, const __bf16* __restrict__ Bm,
               float* __restrict__ Cf, __bf16* __restrict__ Cb,
               const float* __restrict__ bias, const float* __restrict__ res,
               int M, int N, int K)
{
  __shared__ __bf16 sA[128][40];   // row-major [m][k], padded
  __shared__ __bf16 sB[64][40];    // transposed [n][k], padded

  const int tid   = threadIdx.x;
  const int mBase = blockIdx.y * 128;
  const int nBase = blockIdx.x * 64;
  const int lane  = tid & 31;
  const int wid   = tid >> 5;
  const int mWave = (wid & 3) * 32;
  const int nWave = (wid >> 2) * 32;
  const int l16   = lane & 15;
  const int hi    = lane >> 4;
  const int kh    = hi * 8;

  v8f acc[2][2];
  #pragma unroll
  for (int a = 0; a < 2; ++a)
    #pragma unroll
    for (int b = 0; b < 2; ++b)
      #pragma unroll
      for (int r = 0; r < 8; ++r) acc[a][b][r] = 0.0f;

  const int ra  = tid >> 1;          // 0..127 : A row within tile
  const int kca = (tid & 1) * 16;    // 0 or 16
  const int kb  = tid >> 3;          // 0..31  : B k-row
  const int ncb = (tid & 7) * 8;     // 0..56  : B col chunk

  const __bf16* gaBase = A + (size_t)(mBase + ra) * K + kca;
  const __bf16* gbBase = Bm + (size_t)kb * N + nBase + ncb;

  // prologue fetch of tile k0 = 0
  float4 a0 = *(const float4*)(gaBase);
  float4 a1 = *(const float4*)(gaBase + 8);
  float4 b0 = *(const float4*)(gbBase);

  for (int k0 = 0; k0 < K; k0 += 32) {
    // stage current tile into LDS
    *(float4*)&sA[ra][kca]     = a0;
    *(float4*)&sA[ra][kca + 8] = a1;
    const __bf16* btp = (const __bf16*)&b0;
    #pragma unroll
    for (int j = 0; j < 8; ++j) sB[ncb + j][kb] = btp[j];
    __syncthreads();

    // prefetch next tile (overlaps with WMMA below)
    if (k0 + 32 < K) {
      a0 = *(const float4*)(gaBase + (k0 + 32));
      a1 = *(const float4*)(gaBase + (k0 + 32) + 8);
      b0 = *(const float4*)(gbBase + (size_t)(k0 + 32) * N);
    }

    v16bf aF[2], bF[2];
    #pragma unroll
    for (int mi = 0; mi < 2; ++mi) aF[mi] = load_fragA(&sA[mWave + mi * 16 + l16][0], kh);
    #pragma unroll
    for (int ni = 0; ni < 2; ++ni) bF[ni] = load_fragB(&sB[nWave + ni * 16 + l16][0], hi);

    #pragma unroll
    for (int mi = 0; mi < 2; ++mi)
      #pragma unroll
      for (int ni = 0; ni < 2; ++ni)
        acc[mi][ni] = __builtin_amdgcn_wmma_f32_16x16x32_bf16(
            false, aF[mi], false, bF[ni], (short)0, acc[mi][ni], false, false);
    __syncthreads();
  }

  // Epilogue. C/D layout: VGPR r holds row (hi*8 + r), col = l16.
  #pragma unroll
  for (int mi = 0; mi < 2; ++mi)
    #pragma unroll
    for (int ni = 0; ni < 2; ++ni) {
      int col = nBase + nWave + ni * 16 + l16;
      float bv = HAS_BIAS ? bias[col] : 0.0f;
      #pragma unroll
      for (int r = 0; r < 8; ++r) {
        int row = mBase + mWave + mi * 16 + hi * 8 + r;
        float v = acc[mi][ni][r] + bv;
        if (HAS_RES) v += res[(size_t)row * N + col];
        if (RELU) v = fmaxf(v, 0.0f);
        if (BF16OUT) Cb[(size_t)row * N + col] = (__bf16)v;
        else         Cf[(size_t)row * N + col] = v;
      }
    }
}

// ---------------------------------------------------------------------------
// LayerNorm over D=384, one wave per row, bf16 output.
// ---------------------------------------------------------------------------
__global__ __launch_bounds__(256)
void ln_rows(const float* __restrict__ x, const float* __restrict__ g,
             const float* __restrict__ bta, __bf16* __restrict__ out)
{
  const int row  = blockIdx.x * 8 + (threadIdx.x >> 5);
  const int lane = threadIdx.x & 31;
  const float* xr = x + (size_t)row * 384;
  float v[12];
  float s = 0.f;
  #pragma unroll
  for (int i = 0; i < 12; ++i) { v[i] = xr[lane + 32 * i]; s += v[i]; }
  #pragma unroll
  for (int d = 16; d; d >>= 1) s += __shfl_xor(s, d, 32);
  float mu = s * (1.0f / 384.0f);
  float s2 = 0.f;
  #pragma unroll
  for (int i = 0; i < 12; ++i) { float dd = v[i] - mu; s2 += dd * dd; }
  #pragma unroll
  for (int d = 16; d; d >>= 1) s2 += __shfl_xor(s2, d, 32);
  float inv = rsqrtf(s2 * (1.0f / 384.0f) + 1e-5f);
  __bf16* orow = out + (size_t)row * 384;
  #pragma unroll
  for (int i = 0; i < 12; ++i) {
    int c = lane + 32 * i;
    orow[c] = (__bf16)((v[i] - mu) * inv * g[c] + bta[c]);
  }
}

// ---------------------------------------------------------------------------
// Weight conversions
// ---------------------------------------------------------------------------
__global__ void cvt_f32_bf16(const float* __restrict__ in, __bf16* __restrict__ out, int n)
{
  int i = blockIdx.x * 256 + threadIdx.x;
  if (i < n) out[i] = (__bf16)in[i];
}

// wq/wk/wv are [H=6, D=384, HS=64]; fuse into W[c][p*384 + h*64 + s]  (384 x 1152)
__global__ void cvt_qkv(const float* __restrict__ wq, const float* __restrict__ wk,
                        const float* __restrict__ wv, __bf16* __restrict__ out)
{
  int i = blockIdx.x * 256 + threadIdx.x;
  if (i >= 384 * 1152) return;
  int c = i / 1152;
  int t = i - c * 1152;
  int p = t / 384;
  int r = t - p * 384;
  int hh = r >> 6;
  int ss = r & 63;
  const float* w = (p == 0) ? wq : (p == 1) ? wk : wv;
  out[i] = (__bf16)w[((size_t)hh * 384 + c) * 64 + ss];
}

// ---------------------------------------------------------------------------
// Attention: one block (4 waves) per (b,h). T=100 (pad 112 / K-pad 128), HS=64.
// qkv layout per row: [q(384) | k(384) | v(384)], head slice at h*64.
// Output: o[(b*100+t)*384 + h*64 + s]  (heads concatenated), bf16.
// All j-loops fully unrolled (compile-time) so accumulators stay in VGPRs.
// ---------------------------------------------------------------------------
__global__ __launch_bounds__(128)
void attn_wmma(const __bf16* __restrict__ qkv, __bf16* __restrict__ o)
{
  __shared__ __bf16 sQ[112][64];        // 14336 B
  __shared__ __bf16 sK[112][64];        // 14336 B
  __shared__ __bf16 sVt[64][136];       // transposed V [s][t], 17408 B
  __shared__ __bf16 sP[4][16][136];     // per-wave probs, 17408 B

  const int bh  = blockIdx.x;
  const int b   = bh / 6;
  const int h   = bh - b * 6;
  const int tid = threadIdx.x;

  if (tid < 112) {
    int t = tid;
    if (t < 100) {
      const __bf16* base = qkv + (size_t)(b * 100 + t) * 1152 + h * 64;
      #pragma unroll
      for (int j = 0; j < 8; ++j) ((float4*)&sQ[t][0])[j] = ((const float4*)(base))[j];
      #pragma unroll
      for (int j = 0; j < 8; ++j) ((float4*)&sK[t][0])[j] = ((const float4*)(base + 384))[j];
      __bf16 vr[64];
      #pragma unroll
      for (int j = 0; j < 8; ++j) ((float4*)vr)[j] = ((const float4*)(base + 768))[j];
      #pragma unroll
      for (int s = 0; s < 64; ++s) sVt[s][t] = vr[s];
    } else {
      #pragma unroll
      for (int s = 0; s < 64; ++s) { sQ[t][s] = (__bf16)0.f; sK[t][s] = (__bf16)0.f; sVt[s][t] = (__bf16)0.f; }
    }
  } else {
    int t = tid;  // 112..127 : zero-pad V K-dim to 128
    #pragma unroll
    for (int s = 0; s < 64; ++s) sVt[s][t] = (__bf16)0.f;
  }
  __syncthreads();

  const int w    = tid >> 5;
  const int lane = tid & 31;
  const int l16  = lane & 15;
  const int hi   = lane >> 4;
  const int kh   = hi * 8;

  for (int it = 0; it < 2; ++it) {
    const int  i      = w + it * 4;       // row tile (wave w owns tiles w and w+4)
    const bool active = (i < 7);

    if (active) {
      // ---- scores S = q k^T * 8, causal. Compute ALL 7 col tiles (static
      //      unroll); the causal mask zeroes tiles j > i, keeping accS in VGPRs.
      v16bf qF[2];
      #pragma unroll
      for (int ks2 = 0; ks2 < 2; ++ks2) qF[ks2] = load_fragA(&sQ[i * 16 + l16][ks2 * 32], kh);

      v8f accS[7];
      #pragma unroll
      for (int j = 0; j < 7; ++j)
        #pragma unroll
        for (int r = 0; r < 8; ++r) accS[j][r] = 0.f;

      #pragma unroll
      for (int j = 0; j < 7; ++j) {
        #pragma unroll
        for (int ks2 = 0; ks2 < 2; ++ks2) {
          v16bf kF = load_fragB(&sK[j * 16 + l16][ks2 * 32], hi);
          accS[j] = __builtin_amdgcn_wmma_f32_16x16x32_bf16(
              false, qF[ks2], false, kF, (short)0, accS[j], false, false);
        }
      }
      // scale by sqrt(HS)=8 and apply causal + length mask
      #pragma unroll
      for (int j = 0; j < 7; ++j) {
        int col = j * 16 + l16;
        #pragma unroll
        for (int r = 0; r < 8; ++r) {
          int row = i * 16 + hi * 8 + r;
          float sc = accS[j][r] * 8.0f;
          accS[j][r] = ((col <= row) && (col < 100)) ? sc : -1e30f;
        }
      }
      // ---- softmax per row (16-lane reductions; each VGPR r is one row) ----
      float inv[8];
      #pragma unroll
      for (int r = 0; r < 8; ++r) {
        float m = -1e30f;
        #pragma unroll
        for (int j = 0; j < 7; ++j) m = fmaxf(m, accS[j][r]);
        #pragma unroll
        for (int d = 1; d < 16; d <<= 1) m = fmaxf(m, __shfl_xor(m, d, 32));
        float sum = 0.f;
        #pragma unroll
        for (int j = 0; j < 7; ++j) { float e = __expf(accS[j][r] - m); accS[j][r] = e; sum += e; }
        #pragma unroll
        for (int d = 1; d < 16; d <<= 1) sum += __shfl_xor(sum, d, 32);
        inv[r] = 1.0f / sum;
      }
      // ---- write normalized P (bf16) to per-wave LDS (masked entries are 0) ----
      #pragma unroll
      for (int j = 0; j < 7; ++j) {
        #pragma unroll
        for (int r = 0; r < 8; ++r) {
          sP[w][hi * 8 + r][j * 16 + l16] = (__bf16)(accS[j][r] * inv[r]);
        }
      }
    }
    __syncthreads();

    if (active) {
      // ---- O = P * V  (K padded to 128 with zeros) ----
      v8f accO[4];
      #pragma unroll
      for (int ni = 0; ni < 4; ++ni)
        #pragma unroll
        for (int r = 0; r < 8; ++r) accO[ni][r] = 0.f;

      #pragma unroll
      for (int kt = 0; kt < 4; ++kt) {
        v16bf pF = load_fragA(&sP[w][l16][kt * 32], kh);
        #pragma unroll
        for (int ni = 0; ni < 4; ++ni) {
          v16bf vF = load_fragB(&sVt[ni * 16 + l16][kt * 32], hi);
          accO[ni] = __builtin_amdgcn_wmma_f32_16x16x32_bf16(
              false, pF, false, vF, (short)0, accO[ni], false, false);
        }
      }
      #pragma unroll
      for (int ni = 0; ni < 4; ++ni) {
        int ss = h * 64 + ni * 16 + l16;
        #pragma unroll
        for (int r = 0; r < 8; ++r) {
          int t = i * 16 + hi * 8 + r;
          if (t < 100) o[(size_t)(b * 100 + t) * 384 + ss] = (__bf16)accO[ni][r];
        }
      }
    }
    __syncthreads();
  }
}

// ---------------------------------------------------------------------------
// Driver
// ---------------------------------------------------------------------------
extern "C" void kernel_launch(void* const* d_in, const int* in_sizes, int n_in,
                              void* d_out, int out_size, void* d_ws, size_t ws_size,
                              hipStream_t stream)
{
  (void)in_sizes; (void)n_in; (void)out_size; (void)ws_size;
  const float* x     = (const float*)d_in[0];
  const float* wq    = (const float*)d_in[1];
  const float* wk    = (const float*)d_in[2];
  const float* wv    = (const float*)d_in[3];
  const float* wproj = (const float*)d_in[4];
  const float* bproj = (const float*)d_in[5];
  const float* w1    = (const float*)d_in[6];
  const float* b1    = (const float*)d_in[7];
  const float* w2    = (const float*)d_in[8];
  const float* b2    = (const float*)d_in[9];
  const float* g1    = (const float*)d_in[10];
  const float* beta1 = (const float*)d_in[11];
  const float* g2    = (const float*)d_in[12];
  const float* beta2 = (const float*)d_in[13];
  float* out = (float*)d_out;

  const int M = 512 * 100;  // 51200 rows, divisible by 128
  char* ws = (char*)d_ws;
  size_t off = 0;
  auto take = [&](size_t bytes) -> char* {
    char* p = ws + off;
    off += (bytes + 255) & ~(size_t)255;
    return p;
  };

  __bf16* wqkv_b  = (__bf16*)take((size_t)384 * 1152 * 2);
  __bf16* wproj_b = (__bf16*)take((size_t)384 * 384 * 2);
  __bf16* w1_b    = (__bf16*)take((size_t)384 * 1536 * 2);
  __bf16* w2_b    = (__bf16*)take((size_t)1536 * 384 * 2);
  __bf16* hbuf    = (__bf16*)take((size_t)M * 384 * 2);   // ln1 out; reused as attn O
  __bf16* qkvbuf  = (__bf16*)take((size_t)M * 1152 * 2);  // reused as h2 after attention
  float*  x1      = (float*)take((size_t)M * 384 * 4);
  __bf16* ubuf    = (__bf16*)take((size_t)M * 1536 * 2);
  __bf16* obuf    = hbuf;
  __bf16* h2buf   = qkvbuf;

  // weight conversions (tiny)
  cvt_qkv<<<(384 * 1152 + 255) / 256, 256, 0, stream>>>(wq, wk, wv, wqkv_b);
  cvt_f32_bf16<<<(384 * 384  + 255) / 256, 256, 0, stream>>>(wproj, wproj_b, 384 * 384);
  cvt_f32_bf16<<<(384 * 1536 + 255) / 256, 256, 0, stream>>>(w1, w1_b, 384 * 1536);
  cvt_f32_bf16<<<(1536 * 384 + 255) / 256, 256, 0, stream>>>(w2, w2_b, 1536 * 384);

  // LN1 -> h (bf16)
  ln_rows<<<M / 8, 256, 0, stream>>>(x, g1, beta1, hbuf);
  // QKV: [M,384] x [384,1152] -> bf16
  gemm_wmma<false, true, false, false><<<dim3(1152 / 64, M / 128), 256, 0, stream>>>(
      hbuf, wqkv_b, nullptr, qkvbuf, nullptr, nullptr, M, 1152, 384);
  // attention per (b,h)
  attn_wmma<<<512 * 6, 128, 0, stream>>>(qkvbuf, obuf);
  // proj: x1 = x + O @ wproj + bproj   (f32)
  gemm_wmma<false, false, true, true><<<dim3(384 / 64, M / 128), 256, 0, stream>>>(
      obuf, wproj_b, x1, nullptr, bproj, x, M, 384, 384);
  // LN2 -> h2 (bf16)
  ln_rows<<<M / 8, 256, 0, stream>>>(x1, g2, beta2, h2buf);
  // FFN1: relu(h2 @ w1 + b1) -> bf16
  gemm_wmma<true, true, true, false><<<dim3(1536 / 64, M / 128), 256, 0, stream>>>(
      h2buf, w1_b, nullptr, ubuf, b1, nullptr, M, 1536, 384);
  // FFN2: out = x1 + u @ w2 + b2  (f32)
  gemm_wmma<false, false, true, true><<<dim3(384 / 64, M / 128), 256, 0, stream>>>(
      ubuf, w2_b, out, nullptr, b2, x1, M, 384, 1536);
}
